// RWKVBlock_87076166960152
// MI455X (gfx1250) — matrix operand order repaired and checked by reading, not compiled, not run on hardware
//
#include <hip/hip_runtime.h>
#include <hip/hip_bf16.h>
#include <stdint.h>

typedef __attribute__((ext_vector_type(16))) __bf16 v16bf;
typedef __attribute__((ext_vector_type(8)))  float  v8f;
typedef int v4i __attribute__((vector_size(16)));
typedef __attribute__((address_space(1))) v4i* global_v4i_ptr;
typedef __attribute__((address_space(3))) v4i* lds_v4i_ptr;

__device__ __forceinline__ uint16_t f32_to_bf16(float f) {
  uint32_t u = __float_as_uint(f);
  u += 0x7FFFu + ((u >> 16) & 1u);          // round-to-nearest-even
  return (uint16_t)(u >> 16);
}

__device__ __forceinline__ float sigmoidf_(float x) {
  return 1.0f / (1.0f + __expf(-x));
}

// ---------------------------------------------------------------------------
// Weight prep: W (K x N, f32, row-major)  ->  Wt (N x K, bf16, row-major).
// N-major bf16 layout makes each WMMA B-fragment two 16B k-contiguous reads per lane.
// ---------------------------------------------------------------------------
__global__ void rwkv_transpose_bf16(const float* __restrict__ W,
                                    uint16_t* __restrict__ Wt,
                                    int K, int N) {
  __shared__ float tile[32][33];
  int kb = blockIdx.x * 32;
  int nb = blockIdx.y * 32;
  int tx = threadIdx.x & 31;
  int ty = threadIdx.x >> 5;  // 0..7
  #pragma unroll
  for (int i = ty; i < 32; i += 8)
    tile[i][tx] = W[(size_t)(kb + i) * N + nb + tx];
  __syncthreads();
  #pragma unroll
  for (int i = ty; i < 32; i += 8)
    Wt[(size_t)(nb + i) * K + kb + tx] = f32_to_bf16(tile[tx][i]);
}

// ---------------------------------------------------------------------------
// Fused rmsnorm + token-shift + time-mix, emitting bf16 GEMM inputs.
// ---------------------------------------------------------------------------
__global__ void rwkv_mix_kernel(const float* __restrict__ x,
                                const float* __restrict__ lnw,
                                const float* __restrict__ tmk,
                                const float* __restrict__ tmv,
                                const float* __restrict__ tmr,
                                uint16_t* __restrict__ xk,
                                uint16_t* __restrict__ xv,
                                uint16_t* __restrict__ xr,
                                int T, int C, int hasV) {
  int tok = blockIdx.x;
  int t = tok % T;
  const float* cur = x + (size_t)tok * C;
  const float* prv = (t == 0) ? nullptr : (cur - C);
  __shared__ float red1[256];
  __shared__ float red2[256];
  float s1 = 0.f, s2 = 0.f;
  for (int c = threadIdx.x; c < C; c += blockDim.x) {
    float a = cur[c]; s1 += a * a;
    if (prv) { float p = prv[c]; s2 += p * p; }
  }
  red1[threadIdx.x] = s1;
  red2[threadIdx.x] = s2;
  __syncthreads();
  for (int s = blockDim.x >> 1; s > 0; s >>= 1) {
    if ((int)threadIdx.x < s) {
      red1[threadIdx.x] += red1[threadIdx.x + s];
      red2[threadIdx.x] += red2[threadIdx.x + s];
    }
    __syncthreads();
  }
  float inv1 = rsqrtf(red1[0] / (float)C + 1e-6f);
  float inv2 = prv ? rsqrtf(red2[0] / (float)C + 1e-6f) : 0.f;
  size_t base = (size_t)tok * C;
  for (int c = threadIdx.x; c < C; c += blockDim.x) {
    float w  = lnw[c];
    float h  = cur[c] * inv1 * w;
    float sh = prv ? (prv[c] * inv2 * w) : 0.f;
    float mk = tmk[c];
    xk[base + c] = f32_to_bf16(mk * h + (1.f - mk) * sh);
    if (hasV) {
      float mv = tmv[c];
      xv[base + c] = f32_to_bf16(mv * h + (1.f - mv) * sh);
    }
    float mr = tmr[c];
    xr[base + c] = f32_to_bf16(mr * h + (1.f - mr) * sh);
  }
}

// ---------------------------------------------------------------------------
// bf16 WMMA GEMM:  acc = A(MxK) @ Wt(NxK)^T  with fp32 accumulation.
// Block tile 128(M) x 64(N), K-step 64; 4 waves; wave = 32 rows x 64 cols
// (2 row-groups x 4 col-groups = 8 accumulators, 16 WMMA per K-stage).
// B tile staged in LDS via async global->LDS DMA (ASYNCcnt) when available.
// LDS row stride 36 dwords -> conflict-free b128 fragment reads.
// Epilogues: 0 f32 | 1 sigmoid f32 | 2 resid+acc f32 | 3 relu^2 bf16
//            4 resid + gate*acc f32
// ---------------------------------------------------------------------------
#define GEMM_HAS_ASYNC \
  (__has_builtin(__builtin_amdgcn_global_load_async_to_lds_b128) && \
   __has_builtin(__builtin_amdgcn_s_wait_asynccnt))

__global__ __launch_bounds__(128)
void rwkv_gemm_bf16(const uint16_t* __restrict__ A,
                    const uint16_t* __restrict__ Wt,
                    int M, int N, int K,
                    float* __restrict__ outF,
                    uint16_t* __restrict__ outB,
                    const float* __restrict__ resid,
                    const float* __restrict__ gate,
                    int epilogue) {
  __shared__ uint32_t ldsB[64 * 36];       // 64 cols x 64 K bf16 (+pad)
  const int lane = threadIdx.x & 31;
  const int wv   = threadIdx.x >> 5;       // wave 0..3
  const int m0   = blockIdx.x * 128;
  const int n0   = blockIdx.y * 64;
  const int row  = lane & 15;              // row (A) / col (B/C) within 16
  const int hf   = lane >> 4;              // half-wave select

  v8f acc[2][4] = {};

  const uint16_t* aRow0 = A + (size_t)(m0 + wv * 32 + row) * K;
  const uint16_t* aRow1 = aRow0 + (size_t)16 * K;

  // staging assignment: thread -> (output col r, K-half h), 16 dwords each
  const int sr = threadIdx.x >> 1;
  const int sh = threadIdx.x & 1;

  for (int k0 = 0; k0 < K; k0 += 64) {
    // --- stage B tile (64 cols x 64 K bf16 = 8KB) into LDS ---
    {
      const uint32_t* src =
          (const uint32_t*)(Wt + (size_t)(n0 + sr) * K + k0) + sh * 16;
      uint32_t* dst = &ldsB[sr * 36 + sh * 16];
#if GEMM_HAS_ASYNC
      #pragma unroll
      for (int q = 0; q < 4; ++q) {
        __builtin_amdgcn_global_load_async_to_lds_b128(
            (global_v4i_ptr)(src + q * 4),
            (lds_v4i_ptr)(dst + q * 4), 0, 0);
      }
      __builtin_amdgcn_s_wait_asynccnt(0);
#else
      #pragma unroll
      for (int q = 0; q < 4; ++q)
        *(uint4*)(dst + q * 4) = *(const uint4*)(src + q * 4);
#endif
    }
    __syncthreads();

    #pragma unroll
    for (int c = 0; c < 2; ++c) {          // two 32-wide K chunks
      const int kd = (k0 >> 1) + c * 16;   // dword offset into A rows
      // A fragments for both row groups (16x32 bf16 each), straight from global
      union { uint4 u[2]; v16bf v; } a0, a1;
      a0.u[0] = *(const uint4*)((const uint32_t*)aRow0 + kd + hf * 4);
      a0.u[1] = *(const uint4*)((const uint32_t*)aRow0 + kd + hf * 4 + 8);
      a1.u[0] = *(const uint4*)((const uint32_t*)aRow1 + kd + hf * 4);
      a1.u[1] = *(const uint4*)((const uint32_t*)aRow1 + kd + hf * 4 + 8);
      if (c == 0 && k0 + 64 < K) {
        __builtin_prefetch(aRow0 + k0 + 64, 0, 3);
        __builtin_prefetch(aRow1 + k0 + 64, 0, 3);
      }
      #pragma unroll
      for (int j = 0; j < 4; ++j) {
        union { uint4 u[2]; v16bf v; } bf;
        const uint32_t* pb = &ldsB[(j * 16 + row) * 36 + c * 16 + hf * 4];
        bf.u[0] = *(const uint4*)(pb);
        bf.u[1] = *(const uint4*)(pb + 8);
        acc[0][j] = __builtin_amdgcn_wmma_f32_16x16x32_bf16(
            false, a0.v, false, bf.v, (short)0, acc[0][j], false, false);
        acc[1][j] = __builtin_amdgcn_wmma_f32_16x16x32_bf16(
            false, a1.v, false, bf.v, (short)0, acc[1][j], false, false);
      }
    }
    __syncthreads();
  }

  // --- fused epilogue; C layout: VGPR r -> M = r + 8*hf, N = lane&15 ---
  #pragma unroll
  for (int g = 0; g < 2; ++g) {
    #pragma unroll
    for (int j = 0; j < 4; ++j) {
      #pragma unroll
      for (int rr = 0; rr < 8; ++rr) {
        int m = m0 + wv * 32 + g * 16 + hf * 8 + rr;
        int n = n0 + j * 16 + row;
        size_t o = (size_t)m * N + n;
        float a = acc[g][j][rr];
        switch (epilogue) {
          case 0: outF[o] = a; break;
          case 1: outF[o] = sigmoidf_(a); break;
          case 2: outF[o] = resid[o] + a; break;
          case 3: { float t = a > 0.f ? a : 0.f; outB[o] = f32_to_bf16(t * t); } break;
          default: outF[o] = resid[o] + gate[o] * a; break;
        }
      }
    }
  }
}

// ---------------------------------------------------------------------------
// Serial WKV scan over T, parallel over B*C channels; fuses r*wkv -> bf16.
// ---------------------------------------------------------------------------
__global__ void rwkv_wkv_scan(const float* __restrict__ kk,
                              const float* __restrict__ vv,
                              const float* __restrict__ rr,
                              const float* __restrict__ td,
                              const float* __restrict__ tf,
                              uint16_t* __restrict__ rw,
                              int Bn, int T, int C) {
  int id = blockIdx.x * blockDim.x + threadIdx.x;
  if (id >= Bn * C) return;
  int b = id / C, c = id % C;
  float w  = -__expf(td[c]);
  float ew = __expf(fminf(fmaxf(w, -20.f), 20.f));
  float u  = tf[c];
  float aa = 0.f, bb = 0.f;
  size_t base = (size_t)b * T * C + c;
  for (int t = 0; t < T; ++t) {
    size_t idx = base + (size_t)t * C;
    float kt = kk[idx], vt = vv[idx];
    float ekt = __expf(fminf(fmaxf(u + kt, -20.f), 20.f));
    float o = (aa + ekt * vt) / (bb + ekt + 1e-8f);
    float ek = __expf(fminf(fmaxf(kt, -20.f), 20.f));
    aa = ew * aa + ek * vt;
    bb = ew * bb + ek;
    rw[idx] = f32_to_bf16(rr[idx] * o);
  }
}

// ---------------------------------------------------------------------------
// Orchestration. Workspace plan (peak 320 MB, all write-before-read):
//   [0,26MB)    bf16 transposed weights (persist whole launch)
//   [32,128MB)  region A: 3 x 32MB bf16 activation slots (reused 3x)
//   [128,320MB) region B: 3 x 64MB f32 slots (reused: k -> x2, v+r -> k2)
// ---------------------------------------------------------------------------
extern "C" void kernel_launch(void* const* d_in, const int* in_sizes, int n_in,
                              void* d_out, int out_size, void* d_ws, size_t ws_size,
                              hipStream_t stream) {
  (void)in_sizes; (void)n_in; (void)out_size; (void)ws_size;
  const int Bn = 4, T = 4096, C = 1024, I = 4096;
  const int M = Bn * T;

  const float* x         = (const float*)d_in[0];
  const float* ln1_w     = (const float*)d_in[1];
  const float* ln2_w     = (const float*)d_in[2];
  const float* t_decay   = (const float*)d_in[3];
  const float* t_first   = (const float*)d_in[4];
  const float* tmk       = (const float*)d_in[5];
  const float* tmv       = (const float*)d_in[6];
  const float* tmr       = (const float*)d_in[7];
  const float* W_key     = (const float*)d_in[8];
  const float* W_value   = (const float*)d_in[9];
  const float* W_recept  = (const float*)d_in[10];
  const float* W_out     = (const float*)d_in[11];
  const float* tmk2      = (const float*)d_in[12];
  const float* tmr2      = (const float*)d_in[13];
  const float* W_key2    = (const float*)d_in[14];
  const float* W_value2  = (const float*)d_in[15];
  const float* W_recept2 = (const float*)d_in[16];
  float* out = (float*)d_out;

  char* ws = (char*)d_ws;
  const size_t MB = 1ull << 20;
  uint16_t* Wk_t   = (uint16_t*)(ws + 0 * MB);
  uint16_t* Wv_t   = (uint16_t*)(ws + 2 * MB);
  uint16_t* Wr_t   = (uint16_t*)(ws + 4 * MB);
  uint16_t* Wo_t   = (uint16_t*)(ws + 6 * MB);
  uint16_t* Wr2_t  = (uint16_t*)(ws + 8 * MB);
  uint16_t* Wk2_t  = (uint16_t*)(ws + 10 * MB);   // 8 MB (I x C)
  uint16_t* Wv2_t  = (uint16_t*)(ws + 18 * MB);   // 8 MB (C x I)
  uint16_t* xk_bf  = (uint16_t*)(ws + 32 * MB);
  uint16_t* xv_bf  = (uint16_t*)(ws + 64 * MB);
  uint16_t* xr_bf  = (uint16_t*)(ws + 96 * MB);
  float*    kf     = (float*)(ws + 128 * MB);
  float*    vf     = (float*)(ws + 192 * MB);
  float*    rf     = (float*)(ws + 256 * MB);
  uint16_t* rw_bf  = (uint16_t*)(ws + 32 * MB);   // reuse xk slot
  float*    x2     = (float*)(ws + 128 * MB);     // reuse k slot
  uint16_t* xk2_bf = (uint16_t*)(ws + 64 * MB);   // reuse xv slot
  uint16_t* xr2_bf = (uint16_t*)(ws + 96 * MB);   // reuse xr slot
  uint16_t* k2_bf  = (uint16_t*)(ws + 192 * MB);  // 128 MB, reuse v+r slots
  float*    r2f    = (float*)(ws + 32 * MB);      // 64 MB, reuse xk+xv slots

  dim3 tb(256);
  dim3 gb(128);

  // ---- weight prep (bf16, N-major) ----
  rwkv_transpose_bf16<<<dim3(C/32, C/32), tb, 0, stream>>>(W_key,     Wk_t,  C, C);
  rwkv_transpose_bf16<<<dim3(C/32, C/32), tb, 0, stream>>>(W_value,   Wv_t,  C, C);
  rwkv_transpose_bf16<<<dim3(C/32, C/32), tb, 0, stream>>>(W_recept,  Wr_t,  C, C);
  rwkv_transpose_bf16<<<dim3(C/32, C/32), tb, 0, stream>>>(W_out,     Wo_t,  C, C);
  rwkv_transpose_bf16<<<dim3(C/32, C/32), tb, 0, stream>>>(W_recept2, Wr2_t, C, C);
  rwkv_transpose_bf16<<<dim3(C/32, I/32), tb, 0, stream>>>(W_key2,    Wk2_t, C, I);
  rwkv_transpose_bf16<<<dim3(I/32, C/32), tb, 0, stream>>>(W_value2,  Wv2_t, I, C);

  // ---- time mixing ----
  rwkv_mix_kernel<<<dim3(M), tb, 0, stream>>>(x, ln1_w, tmk, tmv, tmr,
                                              xk_bf, xv_bf, xr_bf, T, C, 1);
  rwkv_gemm_bf16<<<dim3(M/128, C/64), gb, 0, stream>>>(xk_bf, Wk_t, M, C, C,
                                                       kf, nullptr, nullptr, nullptr, 0);
  rwkv_gemm_bf16<<<dim3(M/128, C/64), gb, 0, stream>>>(xv_bf, Wv_t, M, C, C,
                                                       vf, nullptr, nullptr, nullptr, 0);
  rwkv_gemm_bf16<<<dim3(M/128, C/64), gb, 0, stream>>>(xr_bf, Wr_t, M, C, C,
                                                       rf, nullptr, nullptr, nullptr, 1);

  rwkv_wkv_scan<<<dim3((Bn * C) / 64), dim3(64), 0, stream>>>(
      kf, vf, rf, t_decay, t_first, rw_bf, Bn, T, C);

  rwkv_gemm_bf16<<<dim3(M/128, C/64), gb, 0, stream>>>(rw_bf, Wo_t, M, C, C,
                                                       x2, nullptr, x, nullptr, 2);

  // ---- channel mixing ----
  rwkv_mix_kernel<<<dim3(M), tb, 0, stream>>>(x2, ln2_w, tmk2, nullptr, tmr2,
                                              xk2_bf, nullptr, xr2_bf, T, C, 0);
  rwkv_gemm_bf16<<<dim3(M/128, I/64), gb, 0, stream>>>(xk2_bf, Wk2_t, M, I, C,
                                                       nullptr, k2_bf, nullptr, nullptr, 3);
  rwkv_gemm_bf16<<<dim3(M/128, C/64), gb, 0, stream>>>(xr2_bf, Wr2_t, M, C, C,
                                                       r2f, nullptr, nullptr, nullptr, 1);
  rwkv_gemm_bf16<<<dim3(M/128, C/64), gb, 0, stream>>>(k2_bf, Wv2_t, M, C, I,
                                                       out, nullptr, x2, r2f, 4);
}
